// Qwen3MoeLayer_61607010894657
// MI455X (gfx1250) — compile-verified, matrix operand order checked by reading, not run on hardware
//
#include <hip/hip_runtime.h>
#include <hip/hip_bf16.h>

// ---------------- dims (fixed by the reference) ----------------
#define TT 2048
#define HH 1024
#define EE 64
#define II 768
#define KK 8

typedef __attribute__((ext_vector_type(16))) __bf16 v16bf;
typedef __attribute__((ext_vector_type(8)))  __bf16 v8bf;
typedef __attribute__((ext_vector_type(4)))  __bf16 v4bf;
typedef __attribute__((ext_vector_type(8)))  float  v8f;

// LDS strides (bf16 elements), 16B-aligned rows
#define XS 1032
#define HS 776
#define XS_BYTES (32 * XS * 2)          // 66048
#define HS_BYTES (32 * HS * 2)          // 49664
#define SMEM_BYTES (XS_BYTES + HS_BYTES + 32 * 4 + 32 * 4)

// ---------------- small wave32 helpers ----------------
static __device__ __forceinline__ float wave_sum(float v) {
#pragma unroll
    for (int m = 16; m; m >>= 1) v += __shfl_xor(v, m, 32);
    return v;
}
static __device__ __forceinline__ float wave_max(float v) {
#pragma unroll
    for (int m = 16; m; m >>= 1) v = fmaxf(v, __shfl_xor(v, m, 32));
    return v;
}

// 16x32 bf16 A-fragment slice from LDS (ISA 7.12.2 layout):
// lane row = lane&15 (caller picks base row), half = lane>>4;
// v16bf idx -> K = (idx>=8?16:0) + half*8 + (idx&7)
static __device__ __forceinline__ v16bf ldsA(const __bf16* row, int kbase, int half) {
    const v8bf* p0 = (const v8bf*)(row + kbase + half * 8);
    const v8bf* p1 = (const v8bf*)(row + kbase + 16 + half * 8);
    v8bf a = *p0;
    v8bf b = *p1;
    return __builtin_shufflevector(a, b, 0, 1, 2, 3, 4, 5, 6, 7, 8, 9, 10, 11, 12, 13, 14, 15);
}

// Raw fp32 weight slice for one B-fragment column (16 floats per lane).
struct Raw { float4 a, b, c, d; };
static __device__ __forceinline__ Raw ldraw(const float* rowbase, int half) {
    const float4* p = (const float4*)rowbase;
    Raw r;
    r.a = p[half * 2 + 0];
    r.b = p[half * 2 + 1];
    r.c = p[4 + half * 2 + 0];
    r.d = p[4 + half * 2 + 1];
    return r;
}
static __device__ __forceinline__ v16bf cvtraw(Raw q) {
    v16bf r;
    r[0]  = (__bf16)q.a.x; r[1]  = (__bf16)q.a.y; r[2]  = (__bf16)q.a.z; r[3]  = (__bf16)q.a.w;
    r[4]  = (__bf16)q.b.x; r[5]  = (__bf16)q.b.y; r[6]  = (__bf16)q.b.z; r[7]  = (__bf16)q.b.w;
    r[8]  = (__bf16)q.c.x; r[9]  = (__bf16)q.c.y; r[10] = (__bf16)q.c.z; r[11] = (__bf16)q.c.w;
    r[12] = (__bf16)q.d.x; r[13] = (__bf16)q.d.y; r[14] = (__bf16)q.d.z; r[15] = (__bf16)q.d.w;
    return r;
}

#define WMMA_BF16(A, B, C) \
    __builtin_amdgcn_wmma_f32_16x16x32_bf16(false, (A), false, (B), (short)0, (C), false, false)

// ---------------- kernel 0: zero output + expert counts ----------------
__global__ void moe_zero_kernel(float* __restrict__ out, int n, int* __restrict__ counts) {
    int i = blockIdx.x * blockDim.x + threadIdx.x;
    int stride = gridDim.x * blockDim.x;
    for (int j = i; j < n; j += stride) out[j] = 0.0f;
    if (i < EE) counts[i] = 0;
}

// ---------------- kernel 1: router (one wave32 per token) ----------------
__global__ void __launch_bounds__(256)
moe_router_kernel(const float* __restrict__ x, const float* __restrict__ wr,
                  int* __restrict__ counts, int* __restrict__ tok_ids,
                  float* __restrict__ tok_w) {
    int t    = blockIdx.x * 8 + (threadIdx.x >> 5);
    int lane = threadIdx.x & 31;
    if (t >= TT) return;
    const float* xrow = x + (size_t)t * HH;

    float l0 = 0.f, l1 = 0.f;   // lane holds experts (lane) and (lane+32)
    for (int e = 0; e < EE; ++e) {
        const float* w = wr + (size_t)e * HH;
        float p = 0.f;
        for (int h = lane; h < HH; h += 32) p += xrow[h] * w[h];
        p = wave_sum(p);
        if ((e & 31) == lane) { if (e < 32) l0 = p; else l1 = p; }
    }
    float mx = wave_max(fmaxf(l0, l1));
    float p0 = __expf(l0 - mx), p1 = __expf(l1 - mx);
    float s = wave_sum(p0 + p1);
    p0 /= s; p1 /= s;

    float cp[KK]; int ce[KK]; float wsum = 0.f;
#pragma unroll
    for (int k = 0; k < KK; ++k) {
        float bv; int bi;
        if (p0 >= p1) { bv = p0; bi = lane; } else { bv = p1; bi = lane + 32; }
#pragma unroll
        for (int m = 16; m; m >>= 1) {
            float ov = __shfl_xor(bv, m, 32);
            int   oi = __shfl_xor(bi, m, 32);
            if (ov > bv || (ov == bv && oi < bi)) { bv = ov; bi = oi; }
        }
        cp[k] = bv; ce[k] = bi; wsum += bv;
        if (bi == lane) p0 = -1.f; else if (bi == lane + 32) p1 = -1.f;
    }
#pragma unroll
    for (int k = 0; k < KK; ++k) {
        if (lane == k) {
            int e = ce[k];
            int slot = atomicAdd(&counts[e], 1);
            tok_ids[e * TT + slot] = t;
            tok_w[e * TT + slot]   = cp[k] / wsum;
        }
    }
}

// ------- kernel 2: grouped expert SwiGLU GEMMs, M=32 token tiles, WMMA bf16 -------
__global__ void __launch_bounds__(256)
moe_expert_kernel(const float* __restrict__ x,  const float* __restrict__ wg,
                  const float* __restrict__ wu, const float* __restrict__ wd,
                  const int* __restrict__ counts, const int* __restrict__ tok_ids,
                  const float* __restrict__ tok_w, float* __restrict__ out) {
    extern __shared__ __align__(16) char smem[];
    __bf16* xs     = (__bf16*)smem;                               // 32 x XS bf16 (x tile)
    __bf16* hs     = (__bf16*)(smem + XS_BYTES);                  // 32 x HS bf16 (h tile)
    int*    toks_s = (int*)  (smem + XS_BYTES + HS_BYTES);        // 32 ids
    float*  wts_s  = (float*)(smem + XS_BYTES + HS_BYTES + 128);  // 32 weights

    int e    = blockIdx.x >> 6;     // expert
    int tile = blockIdx.x & 63;     // 32-token tile within expert
    int cnt  = counts[e];
    int base = tile * 32;
    if (base >= cnt) return;
    int mcnt = min(32, cnt - base);

    int tid  = threadIdx.x;
    int lane = tid & 31;
    int wv   = tid >> 5;
    int n    = lane & 15;
    int half = lane >> 4;

    if (tid < 32) {
        if (tid < mcnt) {
            toks_s[tid] = tok_ids[e * TT + base + tid];
            wts_s[tid]  = tok_w[e * TT + base + tid];
        } else {
            toks_s[tid] = 0;
            wts_s[tid]  = 0.f;
        }
    }
    __syncthreads();

    // stage 32 x rows -> LDS bf16: thread (row = tid>>3, 128-col chunk), b64 stores
    {
        int r  = tid >> 3;
        int c0 = (tid & 7) * 128;
        const float4* xp4 = (const float4*)(x + (size_t)toks_s[r] * HH + c0);
#pragma unroll 8
        for (int j = 0; j < 32; ++j) {
            float4 q = xp4[j];
            v4bf t;
            t[0] = (__bf16)q.x; t[1] = (__bf16)q.y;
            t[2] = (__bf16)q.z; t[3] = (__bf16)q.w;
            *(v4bf*)(xs + r * XS + c0 + j * 4) = t;
        }
    }
    __syncthreads();

    // ---- stage A: g/u = x @ Wg^T / Wu^T, h = silu(g)*u  (48 I-tiles / 8 waves) ----
    for (int it = wv; it < II / 16; it += 8) {
        int i0 = it * 16;
        v8f cg0 = {}, cg1 = {}, cu0 = {}, cu1 = {};
        const float* gB = wg + ((size_t)e * II + i0 + n) * HH;
        const float* uB = wu + ((size_t)e * II + i0 + n) * HH;
        // ping-pong double buffer: slice0 = kk, slice1 = kk+32
        Raw rg0 = ldraw(gB + 0,  half), ru0 = ldraw(uB + 0,  half);
        Raw rg1 = ldraw(gB + 32, half), ru1 = ldraw(uB + 32, half);
        for (int kk = 0; kk < HH; kk += 64) {
            if ((kk & 255) == 0) {
                __builtin_prefetch(gB + kk + 1024, 0, 3);
                __builtin_prefetch(uB + kk + 1024, 0, 3);
            }
            // consume buffer 0 (K=kk), refill for K=kk+64
            v16bf bg = cvtraw(rg0);
            v16bf bu = cvtraw(ru0);
            {
                int kn = (kk + 64 < HH) ? kk + 64 : 0;
                rg0 = ldraw(gB + kn, half);
                ru0 = ldraw(uB + kn, half);
            }
            v16bf a0 = ldsA(xs + n * XS, kk, half);
            v16bf a1 = ldsA(xs + (n + 16) * XS, kk, half);
            cg0 = WMMA_BF16(a0, bg, cg0);
            cg1 = WMMA_BF16(a1, bg, cg1);
            cu0 = WMMA_BF16(a0, bu, cu0);
            cu1 = WMMA_BF16(a1, bu, cu1);
            // consume buffer 1 (K=kk+32), refill for K=kk+96
            v16bf bg2 = cvtraw(rg1);
            v16bf bu2 = cvtraw(ru1);
            {
                int kn = (kk + 96 < HH) ? kk + 96 : 32;
                rg1 = ldraw(gB + kn, half);
                ru1 = ldraw(uB + kn, half);
            }
            a0 = ldsA(xs + n * XS, kk + 32, half);
            a1 = ldsA(xs + (n + 16) * XS, kk + 32, half);
            cg0 = WMMA_BF16(a0, bg2, cg0);
            cg1 = WMMA_BF16(a1, bg2, cg1);
            cu0 = WMMA_BF16(a0, bu2, cu0);
            cu1 = WMMA_BF16(a1, bu2, cu1);
        }
#pragma unroll
        for (int r = 0; r < 8; ++r) {
            int m0 = r + 8 * half;        // C/D layout: m = reg + 8*lane_half
            int m1 = 16 + m0;
            float g0 = cg0[r], u0 = cu0[r];
            float g1 = cg1[r], u1 = cu1[r];
            hs[m0 * HS + i0 + n] = (__bf16)((g0 / (1.f + __expf(-g0))) * u0);
            hs[m1 * HS + i0 + n] = (__bf16)((g1 / (1.f + __expf(-g1))) * u1);
        }
    }
    __syncthreads();

    // ---- stage B: y = h @ Wd^T, scale by route weight, atomic combine ----
    for (int ot = wv; ot < HH / 16; ot += 8) {
        int h0 = ot * 16;
        v8f c0 = {}, c1 = {};
        const float* dB = wd + ((size_t)e * HH + h0 + n) * II;
        Raw rd0 = ldraw(dB + 0,  half);
        Raw rd1 = ldraw(dB + 32, half);
        for (int kk = 0; kk < II; kk += 64) {
            if ((kk & 255) == 0) __builtin_prefetch(dB + kk + 1024, 0, 3);
            v16bf b = cvtraw(rd0);
            {
                int kn = (kk + 64 < II) ? kk + 64 : 0;
                rd0 = ldraw(dB + kn, half);
            }
            v16bf a0 = ldsA(hs + n * HS, kk, half);
            v16bf a1 = ldsA(hs + (n + 16) * HS, kk, half);
            c0 = WMMA_BF16(a0, b, c0);
            c1 = WMMA_BF16(a1, b, c1);
            v16bf b2 = cvtraw(rd1);
            {
                int kn = (kk + 96 < II) ? kk + 96 : 32;
                rd1 = ldraw(dB + kn, half);
            }
            a0 = ldsA(hs + n * HS, kk + 32, half);
            a1 = ldsA(hs + (n + 16) * HS, kk + 32, half);
            c0 = WMMA_BF16(a0, b2, c0);
            c1 = WMMA_BF16(a1, b2, c1);
        }
#pragma unroll
        for (int r = 0; r < 8; ++r) {
            int m0 = r + 8 * half;
            int m1 = 16 + m0;
            if (m0 < mcnt)
                atomicAdd(&out[(size_t)toks_s[m0] * HH + h0 + n], c0[r] * wts_s[m0]);
            if (m1 < mcnt)
                atomicAdd(&out[(size_t)toks_s[m1] * HH + h0 + n], c1[r] * wts_s[m1]);
        }
    }
}

// ---------------- launch ----------------
extern "C" void kernel_launch(void* const* d_in, const int* in_sizes, int n_in,
                              void* d_out, int out_size, void* d_ws, size_t ws_size,
                              hipStream_t stream) {
    (void)in_sizes; (void)n_in; (void)out_size; (void)ws_size;
    const float* x  = (const float*)d_in[0];   // (T,H)
    const float* wr = (const float*)d_in[1];   // (E,H)
    const float* wg = (const float*)d_in[2];   // (E,I,H)
    const float* wu = (const float*)d_in[3];   // (E,I,H)
    const float* wd = (const float*)d_in[4];   // (E,H,I)
    float* out = (float*)d_out;                // (T,H)

    // workspace layout: counts[E] | tok_ids[E*T] | tok_w[E*T]
    int*   counts  = (int*)d_ws;
    int*   tok_ids = counts + EE;
    float* tok_w   = (float*)(tok_ids + EE * TT);

    moe_zero_kernel<<<2048, 256, 0, stream>>>(out, TT * HH, counts);
    moe_router_kernel<<<TT / 8, 256, 0, stream>>>(x, wr, counts, tok_ids, tok_w);
    moe_expert_kernel<<<EE * (TT / 32), 256, SMEM_BYTES, stream>>>(
        x, wg, wu, wd, counts, tok_ids, tok_w, out);
}
